// TermLevelMPN_39084202393945
// MI455X (gfx1250) — matrix operand order repaired and checked by reading, not compiled, not run on hardware
//
#include <hip/hip_runtime.h>

// ---------------------------------------------------------------------------
// Problem constants (match reference)
// ---------------------------------------------------------------------------
#define NT 4096
#define NS 2048
#define NV 1024
#define HH 256
#define E1C 65536
#define E2C 16384
#define E3C 32768
#define LN_EPS 1e-5f

typedef __attribute__((ext_vector_type(16))) __bf16 v16bf;
typedef __attribute__((ext_vector_type(8)))  float  v8f;

#define DEVINL static __device__ __forceinline__

// fp32 -> bf16 (round-to-nearest-even), kept as raw u16 for storage buffers
DEVINL unsigned short f2bfu(float f) {
  unsigned u = __float_as_uint(f);
  return (unsigned short)((u + 0x7FFFu + ((u >> 16) & 1u)) >> 16);
}

union FragBF {
  v16bf v;
  uint4 u[2];
  unsigned short e[16];
};

// A-fragment (16x32, M x K) in contiguous k order (same as B):
//  lane L: m = L&15, khalf = (L>>4)*16 ; elems = A[m][k0+khalf+0..15]
DEVINL v16bf load_a_frag(const unsigned short* A, int lda, int row0, int k0, int lane) {
  int m = lane & 15, g = lane >> 4;
  const unsigned short* p = A + (size_t)(row0 + m) * lda + k0 + g * 16;
  FragBF t;
  t.u[0] = *(const uint4*)(p);
  t.u[1] = *(const uint4*)(p + 8);
  return t.v;
}

// B-fragment (32x16, K x N) from row-major W[N][K]:
//  lane L: n = L&15, khalf = (L>>4)*16 ; elems = W[n0+n][k0+khalf+0..15]
DEVINL v16bf load_b_frag(const unsigned short* B, int ldb, int n0, int k0, int lane) {
  int n = lane & 15, g = lane >> 4;
  const unsigned short* p = B + (size_t)(n0 + n) * ldb + k0 + g * 16;
  FragBF t;
  t.u[0] = *(const uint4*)(p);
  t.u[1] = *(const uint4*)(p + 8);
  return t.v;
}

DEVINL v8f wmma_bf16(v16bf a, v16bf b, v8f c) {
  return __builtin_amdgcn_wmma_f32_16x16x32_bf16(false, a, false, b, (short)0, c,
                                                 false, false);
}

DEVINL v8f zero8() {
  v8f c;
#pragma unroll
  for (int i = 0; i < 8; ++i) c[i] = 0.0f;
  return c;
}

// Async global -> LDS copy (16B), tracked with ASYNCcnt.
// VDST VGPR = wave-relative LDS byte address (low 32 bits of generic pointer),
// per CDNA5 ISA.
DEVINL void async_g2l_b128(unsigned lds_byte_addr, const void* gaddr) {
  unsigned long long ga = (unsigned long long)(uintptr_t)gaddr;
  asm volatile("global_load_async_to_lds_b128 %0, %1, off"
               :
               : "v"(lds_byte_addr), "v"(ga)
               : "memory");
}

DEVINL void wait_async0() {
#if __has_builtin(__builtin_amdgcn_s_wait_asynccnt)
  __builtin_amdgcn_s_wait_asynccnt(0);
#else
  asm volatile("s_wait_asynccnt 0x0" ::: "memory");
#endif
}

// ---------------------------------------------------------------------------
// K0: zero scratch
// ---------------------------------------------------------------------------
__global__ __launch_bounds__(256) void zero_kernel(float* p, size_t n) {
  size_t i = (size_t)blockIdx.x * blockDim.x + threadIdx.x;
  size_t stride = (size_t)gridDim.x * blockDim.x;
  for (; i < n; i += stride) p[i] = 0.0f;
}

// ---------------------------------------------------------------------------
// K1: per-relation edge scatter-add.  64 threads (one float4 each) per edge.
// ---------------------------------------------------------------------------
__global__ __launch_bounds__(256) void edge_agg_kernel(const float* __restrict__ X,
                                                       const int* __restrict__ gidx,
                                                       const int* __restrict__ sidx,
                                                       float* __restrict__ agg,
                                                       float* __restrict__ cnt, int E) {
  int t = blockIdx.x * blockDim.x + threadIdx.x;
  int e = t >> 6, c = t & 63;
  if (e >= E) return;
  int s = gidx[e], d = sidx[e];
  float4 v = ((const float4*)(X + (size_t)s * HH))[c];
  float* ap = agg + (size_t)d * HH + c * 4;
  atomicAdd(ap + 0, v.x);
  atomicAdd(ap + 1, v.y);
  atomicAdd(ap + 2, v.z);
  atomicAdd(ap + 3, v.w);
  if (c == 0) atomicAdd(cnt + d, 1.0f);
}

// ---------------------------------------------------------------------------
// K2: build A_cat bf16 [NT][1280] = [agg0/cnt0 | agg1/cnt1 | agg2/cnt2 |
//                                    agg3/cnt3 | x_term]
// ---------------------------------------------------------------------------
__global__ __launch_bounds__(256) void build_acat_kernel(const float* __restrict__ agg,
                                                         const float* __restrict__ cnt,
                                                         const float* __restrict__ xterm,
                                                         unsigned short* __restrict__ acat) {
  size_t total = (size_t)NT * 1280;
  size_t i = (size_t)blockIdx.x * blockDim.x + threadIdx.x;
  size_t stride = (size_t)gridDim.x * blockDim.x;
  for (; i < total; i += stride) {
    int m = (int)(i / 1280);
    int k = (int)(i % 1280);
    int sec = k >> 8, kk = k & 255;
    float v;
    if (sec < 4)
      v = agg[(size_t)sec * NT * HH + (size_t)m * HH + kk] /
          fmaxf(cnt[sec * NT + m], 1.0f);
    else
      v = xterm[(size_t)m * HH + kk];
    acat[i] = f2bfu(v);
  }
}

// ---------------------------------------------------------------------------
// K3: weight prep: W_cat bf16 [256][1280], b_sum, bf16 copies of dense Ws
// ---------------------------------------------------------------------------
__global__ __launch_bounds__(256) void prep_weights_kernel(
    const float* Wl0, const float* Wl1, const float* Wl2, const float* Wl3,
    const float* Wr0, const float* Wr1, const float* Wr2, const float* Wr3,
    const float* bl0, const float* bl1, const float* bl2, const float* bl3,
    const float* inw, const float* outw, const float* postw,
    unsigned short* wcat, float* bsum, unsigned short* inwb,
    unsigned short* outwb, unsigned short* postwb) {
  const int NWCAT = 256 * 1280;
  const int NIN = 768 * 256;
  const int NSQ = 256 * 256;
  const int total = NWCAT + 256 + NIN + NSQ + NSQ;
  int i = blockIdx.x * blockDim.x + threadIdx.x;
  int stride = gridDim.x * blockDim.x;
  for (; i < total; i += stride) {
    if (i < NWCAT) {
      int n = i / 1280, k = i % 1280;
      int sec = k >> 8, kk = k & 255;
      size_t o = (size_t)n * HH + kk;
      float v;
      if (sec == 0) v = Wl0[o];
      else if (sec == 1) v = Wl1[o];
      else if (sec == 2) v = Wl2[o];
      else if (sec == 3) v = Wl3[o];
      else v = Wr0[o] + Wr1[o] + Wr2[o] + Wr3[o];
      wcat[i] = f2bfu(v);
    } else if (i < NWCAT + 256) {
      int j = i - NWCAT;
      bsum[j] = bl0[j] + bl1[j] + bl2[j] + bl3[j];
    } else if (i < NWCAT + 256 + NIN) {
      int j = i - NWCAT - 256;
      inwb[j] = f2bfu(inw[j]);
    } else if (i < NWCAT + 256 + NIN + NSQ) {
      int j = i - NWCAT - 256 - NIN;
      outwb[j] = f2bfu(outw[j]);
    } else {
      int j = i - NWCAT - 256 - NIN - NSQ;
      postwb[j] = f2bfu(postw[j]);
    }
  }
}

// ---------------------------------------------------------------------------
// K4: fused conv GEMM (4096x1280x256) + bias + residual + LayerNorm
//     block = 256 threads = 8 waves; block owns a 16x256 output tile
// ---------------------------------------------------------------------------
__global__ __launch_bounds__(256) void conv_ln_kernel(
    const unsigned short* __restrict__ acat, const unsigned short* __restrict__ wcat,
    const float* __restrict__ bsum, const float* __restrict__ xterm,
    const float* __restrict__ lng, const float* __restrict__ lnb,
    float* __restrict__ termf, unsigned short* __restrict__ termb) {
  __shared__ float tile[16][HH];
  __shared__ float smu[16], srs[16];
  int lane = threadIdx.x & 31, wave = threadIdx.x >> 5;
  int m0 = blockIdx.x * 16;
  int n0 = wave * 32;

  v8f acc0 = zero8(), acc1 = zero8();
  for (int k0 = 0; k0 < 1280; k0 += 32) {
    v16bf a = load_a_frag(acat, 1280, m0, k0, lane);
    v16bf b0 = load_b_frag(wcat, 1280, n0, k0, lane);
    v16bf b1 = load_b_frag(wcat, 1280, n0 + 16, k0, lane);
    acc0 = wmma_bf16(a, b0, acc0);
    acc1 = wmma_bf16(a, b1, acc1);
  }
  int col0 = n0 + (lane & 15), col1 = col0 + 16;
  int rbase = (lane >> 4) * 8;
#pragma unroll
  for (int r = 0; r < 8; ++r) {
    int rl = rbase + r;
    int row = m0 + rl;
    tile[rl][col0] = acc0[r] + bsum[col0] + xterm[(size_t)row * HH + col0];
    tile[rl][col1] = acc1[r] + bsum[col1] + xterm[(size_t)row * HH + col1];
  }
  __syncthreads();
  if (threadIdx.x < 16) {
    int rl = threadIdx.x;
    float s = 0.0f;
    for (int c = 0; c < HH; ++c) s += tile[rl][c];
    float mu = s * (1.0f / HH);
    float q = 0.0f;
    for (int c = 0; c < HH; ++c) {
      float d = tile[rl][c] - mu;
      q += d * d;
    }
    smu[rl] = mu;
    srs[rl] = rsqrtf(q * (1.0f / HH) + LN_EPS);
  }
  __syncthreads();
  int rl = threadIdx.x >> 4;
  int cb = (threadIdx.x & 15) * 16;
  float mu = smu[rl], rs = srs[rl];
#pragma unroll
  for (int i = 0; i < 16; ++i) {
    int c = cb + i;
    float v = (tile[rl][c] - mu) * rs * lng[c] + lnb[c];
    size_t o = (size_t)(m0 + rl) * HH + c;
    termf[o] = v;
    termb[o] = f2bfu(v);
  }
}

// ---------------------------------------------------------------------------
// K5/K7/K8: generic bf16 WMMA GEMM, C = epilogue(A[MxK] @ B[NxK]^T + bias)
//   flags: 1 = relu, 2 = fp32 output, 4 = add fp32 residual (ld = N)
//   block = 256 (8 waves); block tile = 16 x 256; grid = (M/16, ceil(N/256))
// ---------------------------------------------------------------------------
__global__ __launch_bounds__(256) void gemm_bf16_kernel(
    const unsigned short* __restrict__ A, const unsigned short* __restrict__ B,
    const float* __restrict__ bias, const float* __restrict__ resid,
    void* __restrict__ Cout, int M, int N, int K, int flags) {
  int lane = threadIdx.x & 31, wave = threadIdx.x >> 5;
  int m0 = blockIdx.x * 16;
  int n0 = blockIdx.y * 256 + wave * 32;
  if (n0 >= N) return;  // wave-uniform

  v8f acc0 = zero8(), acc1 = zero8();
  for (int k0 = 0; k0 < K; k0 += 32) {
    v16bf a = load_a_frag(A, K, m0, k0, lane);
    v16bf b0 = load_b_frag(B, K, n0, k0, lane);
    v16bf b1 = load_b_frag(B, K, n0 + 16, k0, lane);
    acc0 = wmma_bf16(a, b0, acc0);
    acc1 = wmma_bf16(a, b1, acc1);
  }
  int col0 = n0 + (lane & 15), col1 = col0 + 16;
  float bb0 = bias[col0], bb1 = bias[col1];
  int rbase = (lane >> 4) * 8;
#pragma unroll
  for (int r = 0; r < 8; ++r) {
    int row = m0 + rbase + r;
    float v0 = acc0[r] + bb0;
    float v1 = acc1[r] + bb1;
    if (flags & 1) {
      v0 = fmaxf(v0, 0.0f);
      v1 = fmaxf(v1, 0.0f);
    }
    if (flags & 4) {
      v0 += resid[(size_t)row * N + col0];
      v1 += resid[(size_t)row * N + col1];
    }
    if (flags & 2) {
      ((float*)Cout)[(size_t)row * N + col0] = v0;
      ((float*)Cout)[(size_t)row * N + col1] = v1;
    } else {
      ((unsigned short*)Cout)[(size_t)row * N + col0] = f2bfu(v0);
      ((unsigned short*)Cout)[(size_t)row * N + col1] = f2bfu(v1);
    }
  }
}

// ---------------------------------------------------------------------------
// K6: flash attention.  One wave per (head, 16-query tile); the 8 waves of a
// block share one head (task = blk*8+wave; 32 blocks span one head's 256
// tiles), so K/V tiles are staged ONCE per block into LDS with
// global_load_async_to_lds_b128 (ASYNCcnt), double-buffered so tile i+1
// streams while tile i feeds the WMMAs.
//   qkv bf16 [4096][768]: q at col h*64, k at 256+h*64, v at 512+h*64
//   attn bf16 [4096][256]: head h at cols h*64..h*64+63
//   1/sqrt(64) folded into S before softmax.
// ---------------------------------------------------------------------------
__global__ __launch_bounds__(256) void flash_attn_kernel(
    const unsigned short* __restrict__ qkv, unsigned short* __restrict__ attn) {
  // [buf][ K tile 32x64 (2048 u16) | V tile 32x64 (2048 u16) ]
  __shared__ __attribute__((aligned(16))) unsigned short kvs[2][4096];
  __shared__ __attribute__((aligned(16))) unsigned short Pt[8][16 * 32];
  int tid = threadIdx.x;
  int lane = tid & 31, wave = tid >> 5;
  int task = blockIdx.x * 8 + wave;
  int head = task >> 8;  // uniform across the block
  int q0 = (task & 255) * 16;

  const unsigned short* qbase = qkv + head * 64;  // [row][dim], ld 768

  // staging role: thread t copies 16B of K and 16B of V of the 32x64 tile
  int srow = tid >> 3;         // key row 0..31 within tile
  int scol = (tid & 7) * 8;    // dim 0,8,...,56
  const unsigned short* kgb = qkv + 256 + head * 64 + scol;
  const unsigned short* vgb = qkv + 512 + head * 64 + scol;
  unsigned ldsK0 = (unsigned)(uintptr_t)&kvs[0][srow * 64 + scol];
  unsigned ldsV0 = (unsigned)(uintptr_t)&kvs[0][2048 + srow * 64 + scol];
  unsigned ldsK1 = (unsigned)(uintptr_t)&kvs[1][srow * 64 + scol];
  unsigned ldsV1 = (unsigned)(uintptr_t)&kvs[1][2048 + srow * 64 + scol];

  v16bf Aq0 = load_a_frag(qbase, 768, q0, 0, lane);
  v16bf Aq1 = load_a_frag(qbase, 768, q0, 32, lane);

  float m_r[8], l_r[8];
  v8f O0 = zero8(), O1 = zero8(), O2 = zero8(), O3 = zero8();
#pragma unroll
  for (int r = 0; r < 8; ++r) {
    m_r[r] = -3.0e38f;
    l_r[r] = 0.0f;
  }

  // prologue: stream tile 0 into buffer 0
  async_g2l_b128(ldsK0, kgb + (size_t)srow * 768);
  async_g2l_b128(ldsV0, vgb + (size_t)srow * 768);

  const int NITER = NT / 32;
  for (int it = 0; it < NITER; ++it) {
    int kb = it * 32;
    int buf = it & 1;
    // tile `it` resident + everyone done reading the other buffer
    wait_async0();
    __syncthreads();
    if (it + 1 < NITER) {  // stream tile it+1 into the other buffer
      size_t grow = (size_t)(kb + 32 + srow) * 768;
      async_g2l_b128(buf ? ldsK0 : ldsK1, kgb + grow);
      async_g2l_b128(buf ? ldsV0 : ldsV1, vgb + grow);
    }
    const unsigned short* Kt = &kvs[buf][0];
    const unsigned short* Vt = &kvs[buf][2048];

    // ---- S = Q @ K^T for 32 keys (two 16-key N-tiles), K from LDS ----
    v8f S0 = zero8(), S1 = zero8();
    {
      v16bf b0 = load_b_frag(Kt, 64, 0, 0, lane);
      v16bf b1 = load_b_frag(Kt, 64, 16, 0, lane);
      S0 = wmma_bf16(Aq0, b0, S0);
      S1 = wmma_bf16(Aq0, b1, S1);
      b0 = load_b_frag(Kt, 64, 0, 32, lane);
      b1 = load_b_frag(Kt, 64, 16, 32, lane);
      S0 = wmma_bf16(Aq1, b0, S0);
      S1 = wmma_bf16(Aq1, b1, S1);
    }
    // ---- online softmax over the 32 new columns ----
#pragma unroll
    for (int r = 0; r < 8; ++r) {
      float s0 = S0[r] * 0.125f;
      float s1 = S1[r] * 0.125f;
      float mx = fmaxf(s0, s1);
      mx = fmaxf(mx, __shfl_xor(mx, 1));
      mx = fmaxf(mx, __shfl_xor(mx, 2));
      mx = fmaxf(mx, __shfl_xor(mx, 4));
      mx = fmaxf(mx, __shfl_xor(mx, 8));
      float mnew = fmaxf(m_r[r], mx);
      float alpha = __expf(m_r[r] - mnew);
      float p0 = __expf(s0 - mnew);
      float p1 = __expf(s1 - mnew);
      float rs = p0 + p1;
      rs += __shfl_xor(rs, 1);
      rs += __shfl_xor(rs, 2);
      rs += __shfl_xor(rs, 4);
      rs += __shfl_xor(rs, 8);
      l_r[r] = l_r[r] * alpha + rs;
      m_r[r] = mnew;
      O0[r] *= alpha;
      O1[r] *= alpha;
      O2[r] *= alpha;
      O3[r] *= alpha;
      int rl = r + (lane >> 4) * 8;
      int col = lane & 15;
      Pt[wave][rl * 32 + col] = f2bfu(p0);
      Pt[wave][rl * 32 + col + 16] = f2bfu(p1);
    }
    // per-wave LDS region; DS ops of one wave complete in order
    v16bf Pa = load_a_frag(&Pt[wave][0], 32, 0, 0, lane);

    // ---- O += P @ V : 4 output dim-subtiles of 16, V gathered from LDS ----
    int n = lane & 15;
    int kh = (lane >> 4) * 16;
#pragma unroll
    for (int j = 0; j < 4; ++j) {
      FragBF bv;
      const unsigned short* vp = Vt + j * 16 + n;
#pragma unroll
      for (int i = 0; i < 16; ++i) bv.e[i] = vp[(kh + i) * 64];
      if (j == 0) O0 = wmma_bf16(Pa, bv.v, O0);
      else if (j == 1) O1 = wmma_bf16(Pa, bv.v, O1);
      else if (j == 2) O2 = wmma_bf16(Pa, bv.v, O2);
      else O3 = wmma_bf16(Pa, bv.v, O3);
    }
  }

  // ---- normalize and write back ----
  int col = lane & 15;
#pragma unroll
  for (int r = 0; r < 8; ++r) {
    float inv = 1.0f / fmaxf(l_r[r], 1e-30f);
    int row = q0 + r + (lane >> 4) * 8;
    size_t o = (size_t)row * HH + head * 64 + col;
    attn[o + 0] = f2bfu(O0[r] * inv);
    attn[o + 16] = f2bfu(O1[r] * inv);
    attn[o + 32] = f2bfu(O2[r] * inv);
    attn[o + 48] = f2bfu(O3[r] * inv);
  }
}

// ---------------------------------------------------------------------------
// Workspace layout (bytes)
// ---------------------------------------------------------------------------
#define WS_AGG 0u
#define WS_CNT (WS_AGG + 4u * NT * HH * 4u)
#define WS_ACAT (WS_CNT + 4u * NT * 4u)
#define WS_WCAT (WS_ACAT + (unsigned)NT * 1280u * 2u)
#define WS_BSUM (WS_WCAT + 256u * 1280u * 2u)
#define WS_TERMF (WS_BSUM + 1024u)
#define WS_TERMB (WS_TERMF + (unsigned)NT * HH * 4u)
#define WS_INW (WS_TERMB + (unsigned)NT * HH * 2u)
#define WS_OUTW (WS_INW + 768u * 256u * 2u)
#define WS_POSTW (WS_OUTW + 256u * 256u * 2u)
#define WS_QKV (WS_POSTW + 256u * 256u * 2u)
#define WS_ATTN (WS_QKV + (unsigned)NT * 768u * 2u)
#define WS_PROJ (WS_ATTN + (unsigned)NT * HH * 2u)

extern "C" void kernel_launch(void* const* d_in, const int* in_sizes, int n_in,
                              void* d_out, int out_size, void* d_ws, size_t ws_size,
                              hipStream_t stream) {
  (void)in_sizes; (void)n_in; (void)out_size; (void)ws_size;

  const float* x_term = (const float*)d_in[0];
  const float* x_symbol = (const float*)d_in[1];
  const float* x_var = (const float*)d_in[2];
  const int* ha_src = (const int*)d_in[3];
  const int* ha_dst = (const int*)d_in[4];
  const int* so_src = (const int*)d_in[5];
  const int* so_dst = (const int*)d_in[6];
  const int* vo_src = (const int*)d_in[7];
  const int* vo_dst = (const int*)d_in[8];
  const float* Wl_rha = (const float*)d_in[9];
  const float* bl_rha = (const float*)d_in[10];
  const float* Wr_rha = (const float*)d_in[11];
  const float* Wl_ha = (const float*)d_in[12];
  const float* bl_ha = (const float*)d_in[13];
  const float* Wr_ha = (const float*)d_in[14];
  const float* Wl_rsym = (const float*)d_in[15];
  const float* bl_rsym = (const float*)d_in[16];
  const float* Wr_rsym = (const float*)d_in[17];
  const float* Wl_var = (const float*)d_in[18];
  const float* bl_var = (const float*)d_in[19];
  const float* Wr_var = (const float*)d_in[20];
  const float* ln_g = (const float*)d_in[21];
  const float* ln_b = (const float*)d_in[22];
  const float* in_w = (const float*)d_in[23];
  const float* in_b = (const float*)d_in[24];
  const float* out_w = (const float*)d_in[25];
  const float* out_b = (const float*)d_in[26];
  const float* post_w = (const float*)d_in[27];
  const float* post_b = (const float*)d_in[28];

  char* ws = (char*)d_ws;
  float* agg = (float*)(ws + WS_AGG);
  float* cnt = (float*)(ws + WS_CNT);
  unsigned short* acat = (unsigned short*)(ws + WS_ACAT);
  unsigned short* wcat = (unsigned short*)(ws + WS_WCAT);
  float* bsum = (float*)(ws + WS_BSUM);
  float* termf = (float*)(ws + WS_TERMF);
  unsigned short* termb = (unsigned short*)(ws + WS_TERMB);
  unsigned short* inwb = (unsigned short*)(ws + WS_INW);
  unsigned short* outwb = (unsigned short*)(ws + WS_OUTW);
  unsigned short* postwb = (unsigned short*)(ws + WS_POSTW);
  unsigned short* qkvb = (unsigned short*)(ws + WS_QKV);
  unsigned short* attnb = (unsigned short*)(ws + WS_ATTN);
  unsigned short* projb = (unsigned short*)(ws + WS_PROJ);

  // K0: zero agg + cnt (contiguous)
  zero_kernel<<<2048, 256, 0, stream>>>(agg, (size_t)4 * NT * HH + 4 * NT);

  // K1: 4 relations (gather index, scatter index)
  edge_agg_kernel<<<E1C / 4, 256, 0, stream>>>(x_term, ha_dst, ha_src,
                                               agg + 0 * (size_t)NT * HH, cnt + 0 * NT, E1C);
  edge_agg_kernel<<<E1C / 4, 256, 0, stream>>>(x_term, ha_src, ha_dst,
                                               agg + 1 * (size_t)NT * HH, cnt + 1 * NT, E1C);
  edge_agg_kernel<<<E2C / 4, 256, 0, stream>>>(x_symbol, so_dst, so_src,
                                               agg + 2 * (size_t)NT * HH, cnt + 2 * NT, E2C);
  edge_agg_kernel<<<E3C / 4, 256, 0, stream>>>(x_var, vo_src, vo_dst,
                                               agg + 3 * (size_t)NT * HH, cnt + 3 * NT, E3C);

  // K2/K3: staging
  build_acat_kernel<<<4096, 256, 0, stream>>>(agg, cnt, x_term, acat);
  prep_weights_kernel<<<1024, 256, 0, stream>>>(
      Wl_rha, Wl_ha, Wl_rsym, Wl_var, Wr_rha, Wr_ha, Wr_rsym, Wr_var,
      bl_rha, bl_ha, bl_rsym, bl_var, in_w, out_w, post_w,
      wcat, bsum, inwb, outwb, postwb);

  // K4: conv GEMM + residual + LayerNorm
  conv_ln_kernel<<<NT / 16, 256, 0, stream>>>(acat, wcat, bsum, x_term, ln_g, ln_b,
                                              termf, termb);

  // K5: QKV projection (M=4096, K=256, N=768)
  gemm_bf16_kernel<<<dim3(NT / 16, 3), 256, 0, stream>>>(
      termb, inwb, in_b, nullptr, qkvb, NT, 768, HH, 0);

  // K6: flash attention (4 heads x 256 query tiles, 8 waves/block)
  flash_attn_kernel<<<128, 256, 0, stream>>>(qkvb, attnb);

  // K7: output projection (M=4096, K=256, N=256)
  gemm_bf16_kernel<<<dim3(NT / 16, 1), 256, 0, stream>>>(
      attnb, outwb, out_b, nullptr, projb, NT, HH, HH, 0);

  // K8: post FFN: relu(proj @ post_w^T + post_b) + term_out -> d_out (fp32)
  gemm_bf16_kernel<<<dim3(NT / 16, 1), 256, 0, stream>>>(
      projb, postwb, post_b, termf, d_out, NT, HH, HH, 1 | 2 | 4);
}
